// DeepseekV2MoE_65128884077160
// MI455X (gfx1250) — matrix operand order repaired and checked by reading, not compiled, not run on hardware
//
#include <hip/hip_runtime.h>
#include <hip/hip_bf16.h>
#include <math.h>

// Problem constants (match reference)
#define T_TOK 512
#define H_DIM 2048
#define I_DIM 1408
#define E_NUM 16
#define G_NUM 4
#define EPG   4      // experts per group
#define TOPKG 2
#define TOPK  6
#define RSF   2.5f

typedef __bf16 bf16;
typedef __attribute__((ext_vector_type(16))) __bf16 v16bf;
typedef __attribute__((ext_vector_type(8)))  __bf16 v8bf;
typedef __attribute__((ext_vector_type(8)))  float  v8f;

// ---------------------------------------------------------------------------
// WMMA fragment helpers (layouts per CDNA5 ISA 7.12.2, wave32)
// A 16x32 bf16: lane l holds row (l&15); element i -> K = (l>>4)*8 + (i>>3)*16 + (i&7)
// B 32x16 bf16: lane l holds col (l&15); element i -> K = (l>>4)*8 + (i>>3)*16 + (i&7)
// C/D 16x16 f32: elem r -> row (l>>4)*8 + r, col (l&15)
// ---------------------------------------------------------------------------

// A-operand from a bf16 row (two contiguous 16B chunks per lane)
__device__ __forceinline__ v16bf load_fragA_bf16(const bf16* row, int k0, int half) {
  const bf16* p = row + k0 + half * 8;
  v8bf lo = *(const v8bf*)(p);
  v8bf hi = *(const v8bf*)(p + 16);
  v16bf r;
#pragma unroll
  for (int i = 0; i < 8; ++i) { r[i] = lo[i]; r[8 + i] = hi[i]; }
  return r;
}

// B-operand from an fp32 weight row (two runs of 8 contiguous floats per lane, 16 apart, cvt->bf16)
__device__ __forceinline__ v16bf load_fragB_f32(const float* row, int k0, int half) {
  const float* p = row + k0 + half * 8;
  v16bf r;
#pragma unroll
  for (int i = 0; i < 16; i += 4) {
    float4 f = *(const float4*)(p + (i & 7) + (i >> 3) * 16);
    r[i]     = (__bf16)f.x;
    r[i + 1] = (__bf16)f.y;
    r[i + 2] = (__bf16)f.z;
    r[i + 3] = (__bf16)f.w;
  }
  return r;
}

// ---------------------------------------------------------------------------
// Kernel 0: zero output accumulator, expert counters, and the zero-row pad
// ---------------------------------------------------------------------------
__global__ void moe_init_kernel(float* __restrict__ out, int* __restrict__ cnt,
                                int* __restrict__ zrow_i32) {
  int i = blockIdx.x * blockDim.x + threadIdx.x;
  if (i < T_TOK * H_DIM) out[i] = 0.0f;
  if (i < E_NUM) cnt[i] = 0;
  if (i < H_DIM / 2) zrow_i32[i] = 0;   // 2048 bf16 zeros
}

// ---------------------------------------------------------------------------
// Kernel 1: routing. One block (128 thr) per token.
// ---------------------------------------------------------------------------
__global__ void route_kernel(const float* __restrict__ x,
                             const float* __restrict__ gw,
                             const float* __restrict__ bias,
                             int* __restrict__ cnt,
                             int* __restrict__ list,
                             float* __restrict__ coefL) {
  __shared__ float lg[E_NUM];
  const int t   = blockIdx.x;
  const int tid = threadIdx.x;           // 0..127
  const int e   = tid >> 3;              // 16 experts x 8 lanes
  const int sub = tid & 7;

  const float* xr = x + (size_t)t * H_DIM;
  const float* gr = gw + (size_t)e * H_DIM;
  float p = 0.0f;
  for (int k = sub; k < H_DIM; k += 8) p += xr[k] * gr[k];
  p += __shfl_xor(p, 1);
  p += __shfl_xor(p, 2);
  p += __shfl_xor(p, 4);
  if (sub == 0) lg[e] = p;
  __syncthreads();

  if (tid == 0) {
    float sc[E_NUM], sf[E_NUM];
    for (int i = 0; i < E_NUM; ++i) {
      float s = 1.0f / (1.0f + expf(-lg[i]));
      sc[i] = s;
      sf[i] = s + bias[i];
    }
    // group scores = sum of top-2 biased scores in each group of 4
    float gs[G_NUM];
    for (int g = 0; g < G_NUM; ++g) {
      float m1 = -1e30f, m2 = -1e30f;
      for (int j = 0; j < EPG; ++j) {
        float v = sf[g * EPG + j];
        if (v > m1) { m2 = m1; m1 = v; } else if (v > m2) { m2 = v; }
      }
      gs[g] = m1 + m2;
    }
    int g1 = 0;
    for (int g = 1; g < G_NUM; ++g) if (gs[g] > gs[g1]) g1 = g;
    int g2 = -1;
    for (int g = 0; g < G_NUM; ++g) {
      if (g == g1) continue;
      if (g2 < 0 || gs[g] > gs[g2]) g2 = g;
    }
    // top-6 experts among the 2 allowed groups (by biased score)
    bool chosen[E_NUM];
    for (int i = 0; i < E_NUM; ++i) chosen[i] = false;
    int idxs[TOPK];
    float wsum = 0.0f;
    for (int k = 0; k < TOPK; ++k) {
      int best = -1; float bv = -1e30f;
      for (int i = 0; i < E_NUM; ++i) {
        int g = i >> 2;
        if (g != g1 && g != g2) continue;
        if (chosen[i]) continue;
        if (sf[i] > bv) { bv = sf[i]; best = i; }
      }
      chosen[best] = true;
      idxs[k] = best;
      wsum += sc[best];      // weights use UNbiased sigmoid scores
    }
    float scale = RSF / wsum;
    for (int k = 0; k < TOPK; ++k) {
      int ee  = idxs[k];
      int pos = atomicAdd(&cnt[ee], 1);
      list[ee * T_TOK + pos]  = t;
      coefL[ee * T_TOK + pos] = sc[ee] * scale;
    }
  }
}

// ---------------------------------------------------------------------------
// Kernel 2: fp32 -> bf16 activation conversion
// ---------------------------------------------------------------------------
__global__ void cvt_kernel(const float* __restrict__ x, bf16* __restrict__ xb) {
  int i = blockIdx.x * blockDim.x + threadIdx.x;
  if (i < T_TOK * H_DIM) xb[i] = (bf16)x[i];
}

// ---------------------------------------------------------------------------
// Kernel 3: gate/up grouped GEMM + SiLU*mul -> h (bf16)
// Each wave computes a 64(M) x 16(N) tile for BOTH gate and up projections:
// 8 wmma per 32-wide k-step against one gate-B + one up-B fetch.
// grid: (I/64, T/64 token-groups, E experts), 128 thr (4 waves, 1 N-tile each)
// ---------------------------------------------------------------------------
__global__ void gateup_kernel(const bf16* __restrict__ xb,
                              const float* __restrict__ wg,
                              const float* __restrict__ wu,
                              const int* __restrict__ cnt,
                              const int* __restrict__ list,
                              const bf16* __restrict__ zrow,
                              bf16* __restrict__ hbuf) {
  const int e      = blockIdx.z;
  const int mgroup = blockIdx.y;            // 64 tokens per group
  const int n_e    = cnt[e];
  if (mgroup * 64 >= n_e) return;

  const int wave  = threadIdx.x >> 5;
  const int lane  = threadIdx.x & 31;
  const int ntile = blockIdx.x * 4 + wave;  // 88 tiles over I
  const int ncol  = ntile * 16 + (lane & 15);
  const int half  = lane >> 4;
  const int m     = lane & 15;

  // Gather: pointer per M-subtile; padded rows read the pre-zeroed row.
  const bf16* arow[4];
#pragma unroll
  for (int s = 0; s < 4; ++s) {
    int slot = mgroup * 64 + s * 16 + m;
    int t = (slot < n_e) ? list[e * T_TOK + slot] : -1;
    arow[s] = (t >= 0) ? (xb + (size_t)t * H_DIM) : zrow;
  }

  const float* grow = wg + ((size_t)e * I_DIM + ncol) * H_DIM;
  const float* urow = wu + ((size_t)e * I_DIM + ncol) * H_DIM;

  v8f accg[4] = {};
  v8f accu[4] = {};
  for (int k0 = 0; k0 < H_DIM; k0 += 32) {
    v16bf bg = load_fragB_f32(grow, k0, half);
    v16bf bu = load_fragB_f32(urow, k0, half);
#pragma unroll
    for (int s = 0; s < 4; ++s) {
      v16bf a = load_fragA_bf16(arow[s], k0, half);
      accg[s] = __builtin_amdgcn_wmma_f32_16x16x32_bf16(false, a, false, bg,
                                                        (short)0, accg[s], false, false);
      accu[s] = __builtin_amdgcn_wmma_f32_16x16x32_bf16(false, a, false, bu,
                                                        (short)0, accu[s], false, false);
    }
  }

#pragma unroll
  for (int s = 0; s < 4; ++s) {
#pragma unroll
    for (int r = 0; r < 8; ++r) {
      int srow = mgroup * 64 + s * 16 + half * 8 + r;
      float g = accg[s][r], u = accu[s][r];
      float h = (g / (1.0f + expf(-g))) * u;    // SiLU(gate) * up
      hbuf[((size_t)e * T_TOK + srow) * I_DIM + ncol] = (bf16)h;
    }
  }
}

// ---------------------------------------------------------------------------
// Kernel 4: down grouped GEMM + weighted scatter into out
// Each wave computes a 64(M) x 16(N) tile: 4 wmma per k-step per B fetch.
// grid: (H/64, T/64 token-groups, E experts), 128 thr
// ---------------------------------------------------------------------------
__global__ void down_kernel(const bf16* __restrict__ hbuf,
                            const float* __restrict__ wd,
                            const int* __restrict__ cnt,
                            const int* __restrict__ list,
                            const float* __restrict__ coefL,
                            float* __restrict__ out) {
  const int e      = blockIdx.z;
  const int mgroup = blockIdx.y;
  const int n_e    = cnt[e];
  if (mgroup * 64 >= n_e) return;

  const int wave  = threadIdx.x >> 5;
  const int lane  = threadIdx.x & 31;
  const int ntile = blockIdx.x * 4 + wave;   // 128 tiles over H
  const int ncol  = ntile * 16 + (lane & 15);
  const int half  = lane >> 4;
  const int m     = lane & 15;

  // All rows in a launched group were written by gateup (padded rows are 0).
  const bf16* arow[4];
#pragma unroll
  for (int s = 0; s < 4; ++s)
    arow[s] = hbuf + ((size_t)e * T_TOK + mgroup * 64 + s * 16 + m) * I_DIM;

  const float* brow = wd + ((size_t)e * H_DIM + ncol) * I_DIM;

  v8f acc[4] = {};
  for (int k0 = 0; k0 < I_DIM; k0 += 32) {
    v16bf b = load_fragB_f32(brow, k0, half);
#pragma unroll
    for (int s = 0; s < 4; ++s) {
      v16bf a = load_fragA_bf16(arow[s], k0, half);
      acc[s] = __builtin_amdgcn_wmma_f32_16x16x32_bf16(false, a, false, b,
                                                       (short)0, acc[s], false, false);
    }
  }

#pragma unroll
  for (int s = 0; s < 4; ++s) {
#pragma unroll
    for (int r = 0; r < 8; ++r) {
      int srow = mgroup * 64 + s * 16 + half * 8 + r;
      if (srow < n_e) {
        int   t = list[e * T_TOK + srow];
        float c = coefL[e * T_TOK + srow];
        atomicAdd(out + (size_t)t * H_DIM + ncol, c * acc[s][r]);
      }
    }
  }
}

// ---------------------------------------------------------------------------
// Launcher
// ---------------------------------------------------------------------------
extern "C" void kernel_launch(void* const* d_in, const int* in_sizes, int n_in,
                              void* d_out, int out_size, void* d_ws, size_t ws_size,
                              hipStream_t stream) {
  const float* x    = (const float*)d_in[0];  // [T,H]
  const float* gw   = (const float*)d_in[1];  // [E,H]
  const float* bias = (const float*)d_in[2];  // [E]
  const float* wg   = (const float*)d_in[3];  // [E,I,H]
  const float* wu   = (const float*)d_in[4];  // [E,I,H]
  const float* wd   = (const float*)d_in[5];  // [E,H,I]
  float* out = (float*)d_out;                 // [T,H]

  // Workspace layout (all offsets 256B-aligned)
  char* ws = (char*)d_ws;
  int*   cnt   = (int*)(ws);                         // 16 ints
  int*   list  = (int*)(ws + 256);                   // E*T ints   (32 KB)
  float* coefL = (float*)(ws + 256 + 32768);         // E*T floats (32 KB)
  bf16*  zrow  = (bf16*)(ws + 65792);                // H_DIM bf16 zeros (4 KB)
  bf16*  xb    = (bf16*)(ws + 69888);                // T*H bf16   (2 MB)
  bf16*  hbuf  = (bf16*)(ws + 69888 + (size_t)T_TOK * H_DIM * 2); // E*T*I bf16 (~22 MB)

  const int n_elem = T_TOK * H_DIM;

  moe_init_kernel<<<(n_elem + 255) / 256, 256, 0, stream>>>(out, cnt, (int*)zrow);
  route_kernel<<<T_TOK, 128, 0, stream>>>(x, gw, bias, cnt, list, coefL);
  cvt_kernel<<<(n_elem + 255) / 256, 256, 0, stream>>>(x, xb);
  gateup_kernel<<<dim3(I_DIM / 64, T_TOK / 64, E_NUM), 128, 0, stream>>>(
      xb, wg, wu, cnt, list, zrow, hbuf);
  down_kernel<<<dim3(H_DIM / 64, T_TOK / 64, E_NUM), 128, 0, stream>>>(
      hbuf, wd, cnt, list, coefL, out);
}
